// LanguageModel_7481833029766
// MI455X (gfx1250) — matrix operand order repaired and checked by reading, not compiled, not run on hardware
//
#include <hip/hip_runtime.h>
#include <hip/hip_bf16.h>
#include <math.h>

// Model dims
#define VSZ    32000
#define EDIM   1024
#define HDIM   1024
#define LAYERS 3
#define TSTEPS 256
#define BATCH  32
#define G4H    (4 * HDIM)
#define KDIM   1024   // E == H == 1024: every GEMM in this model reduces over 1024

#define STEP_BLOCKS 16   // persistent recurrence: 16 WGs x 8 waves = 128 waves

typedef __bf16 bf16;
typedef __attribute__((ext_vector_type(16))) __bf16        v16bf;
typedef __attribute__((ext_vector_type(8)))  float         v8f;
typedef __attribute__((ext_vector_type(4)))  unsigned int  v4u;
typedef __attribute__((ext_vector_type(8)))  int           v8i;
typedef __attribute__((ext_vector_type(4)))  int           v4i;

union FragU {
  v16bf v;
  unsigned int u[8];
  uint4 q[2];
};

// ---------------------------------------------------------------------------
// WMMA fragment helpers (CDNA5 16x16x32 bf16, wave32; layouts per ISA 7.12.2)
// ---------------------------------------------------------------------------
__device__ __forceinline__ v16bf load_a_frag(const bf16* __restrict__ base, int lda) {
  const int lane = threadIdx.x & 31;
  const int r    = lane & 15;
  const int koff = (lane >> 4) << 3;  // 0 or 8
  FragU f;
  const char* p = (const char*)(base + (size_t)r * lda);
#pragma unroll
  for (int v = 0; v < 8; ++v) {
    const int k = koff + ((v & 4) << 2) + ((v & 3) << 1);  // +16 for v>=4, K pairs
    f.u[v] = *(const unsigned int*)(p + 2 * k);
  }
  return f.v;
}

__device__ __forceinline__ v16bf load_b_frag(const bf16* __restrict__ base, int ldb) {
  const int lane = threadIdx.x & 31;
  const int n    = lane & 15;
  const int ksel = (lane >> 4) << 3;  // 0 or 8
  FragU f;
  const uint4* p = (const uint4*)(base + (size_t)n * ldb + ksel);  // 16B aligned
  f.q[0] = p[0];
  f.q[1] = p[2];
  return f.v;
}

__device__ __forceinline__ v8f wmma_bf16(v16bf a, v16bf b, v8f c) {
  return __builtin_amdgcn_wmma_f32_16x16x32_bf16(false, a, false, b, (short)0, c,
                                                 false, false);
}

__device__ __forceinline__ float sigmoidf_(float x) {
  return 1.0f / (1.0f + __expf(-x));
}

// ---------------------------------------------------------------------------
// Tensor Data Mover: DMA one 2D bf16 tile [rows x 32] (row-major, stride
// ld_elems) from global into LDS at byte offset lds_off. Tracked by TENSORcnt.
// ---------------------------------------------------------------------------
__device__ __forceinline__ void tdm_load_tile_2d(const bf16* gsrc, unsigned lds_off,
                                                 int rows, int ld_elems) {
  const unsigned long long ga = (unsigned long long)(size_t)gsrc;
  v4u g0;
  g0[0] = 1u;                                                // count=1, user mode
  g0[1] = lds_off;                                           // lds_addr (bytes)
  g0[2] = (unsigned)ga;                                      // global_addr[31:0]
  g0[3] = (unsigned)((ga >> 32) & 0x01FFFFFFu) | (2u << 30); // addr[56:32]|type=2
  v8i g1;
  g1[0] = (int)(1u << 16);                                   // data_size=1 -> 2B
  g1[1] = (int)((unsigned)ld_elems << 16);                   // tensor_dim0 lo
  g1[2] = (int)(((unsigned)ld_elems >> 16) |
                ((unsigned)rows << 16));                     // dim0 hi | dim1 lo
  g1[3] = (int)(((unsigned)rows >> 16) | (32u << 16));       // dim1 hi | tile_dim0
  g1[4] = rows;                                              // tile_dim1
  g1[5] = ld_elems;                                          // dim0_stride lo
  g1[6] = 0;
  g1[7] = 0;
  const v4i z4 = {0, 0, 0, 0};
#if defined(__clang_major__) && (__clang_major__ >= 23)
  const v8i z8 = {0, 0, 0, 0, 0, 0, 0, 0};
  __builtin_amdgcn_tensor_load_to_lds(g0, g1, z4, z4, z8, 0);
#else
  __builtin_amdgcn_tensor_load_to_lds(g0, g1, z4, z4, 0);
#endif
}

// ---------------------------------------------------------------------------
// Elementwise prep kernels
// ---------------------------------------------------------------------------
__global__ void convert_kernel(const float* __restrict__ src,
                               const float* __restrict__ mask,
                               bf16* __restrict__ dst, int hasMask, size_t n) {
  const size_t i = ((size_t)blockIdx.x * blockDim.x + threadIdx.x) * 4;
  if (i >= n) return;
  float4 s = *(const float4*)(src + i);
  if (hasMask) {
    const float4 m = *(const float4*)(mask + i);
    s.x *= m.x; s.y *= m.y; s.z *= m.z; s.w *= m.w;
  }
  dst[i + 0] = (bf16)s.x; dst[i + 1] = (bf16)s.y;
  dst[i + 2] = (bf16)s.z; dst[i + 3] = (bf16)s.w;
}

__global__ void embed_kernel(const int* __restrict__ x, const float* __restrict__ emb,
                             const float* __restrict__ m_emb, bf16* __restrict__ out) {
  const size_t gid  = (size_t)blockIdx.x * blockDim.x + threadIdx.x;
  const size_t flat = gid * 4;
  const int tb = (int)(flat / EDIM);
  const int e  = (int)(flat % EDIM);
  const int b  = tb % BATCH;
  const int tok = x[tb];
  const float4 ev = *(const float4*)(emb + (size_t)tok * EDIM + e);
  const float4 mv = *(const float4*)(m_emb + (size_t)b * EDIM + e);
  bf16* o = out + flat;
  o[0] = (bf16)(ev.x * mv.x); o[1] = (bf16)(ev.y * mv.y);
  o[2] = (bf16)(ev.z * mv.z); o[3] = (bf16)(ev.w * mv.w);
}

// h slot0 = bf16(h0[l]); c = c0[l]; also reset the grid-barrier counter.
__global__ void init_state_kernel(const float* __restrict__ h0,
                                  const float* __restrict__ c0,
                                  bf16* __restrict__ h, float* __restrict__ c,
                                  unsigned* __restrict__ barrier_cnt) {
  const int i = blockIdx.x * blockDim.x + threadIdx.x;  // B*H
  h[i] = (bf16)h0[i];
  c[i] = c0[i];
  if (i == 0) *barrier_cnt = 0;
}

// ---------------------------------------------------------------------------
// Unified tiled GEMM:  C[M,N] = A[M,1024] @ W[N,1024]^T + bias
// WG = 8 waves, 128x128 tile, double-buffered LDS filled by TDM from wave 0.
// ---------------------------------------------------------------------------
__global__ __launch_bounds__(256) void gemm_tiled_kernel(
    const bf16* __restrict__ A, const bf16* __restrict__ W,
    const float* __restrict__ bias0, const float* __restrict__ bias1,
    int twoBias, float* __restrict__ C, int N, int ntStore) {
  __shared__ __align__(16) bf16 As[2][128 * 32];
  __shared__ __align__(16) bf16 Bs[2][128 * 32];

  const int nb   = N >> 7;
  const int mblk = blockIdx.x / nb;
  const int nblk = blockIdx.x % nb;
  const int wave = threadIdx.x >> 5;
  const int wm   = wave >> 2;
  const int wn   = wave & 3;
  const int m0   = (mblk << 7) + (wm << 6);
  const int n0   = (nblk << 7) + (wn << 5);

  const bf16* Ag = A + (size_t)(mblk << 7) * KDIM;
  const bf16* Wg = W + (size_t)(nblk << 7) * KDIM;
  const bool issuer = (wave == 0);

  if (issuer) {
    tdm_load_tile_2d(Ag, (unsigned)(size_t)&As[0][0], 128, KDIM);
    tdm_load_tile_2d(Wg, (unsigned)(size_t)&Bs[0][0], 128, KDIM);
    __builtin_amdgcn_s_wait_tensorcnt(0);
  }
  __syncthreads();

  v8f acc[4][2] = {};
  const int NK = KDIM / 32;
  for (int kb = 0; kb < NK; ++kb) {
    const int buf = kb & 1;
    if (issuer && (kb + 1 < NK)) {
      tdm_load_tile_2d(Ag + (kb + 1) * 32, (unsigned)(size_t)&As[buf ^ 1][0], 128, KDIM);
      tdm_load_tile_2d(Wg + (kb + 1) * 32, (unsigned)(size_t)&Bs[buf ^ 1][0], 128, KDIM);
    }
    v16bf a[4], b[2];
#pragma unroll
    for (int i = 0; i < 4; ++i)
      a[i] = load_a_frag(&As[buf][(wm * 64 + i * 16) * 32], 32);
#pragma unroll
    for (int j = 0; j < 2; ++j)
      b[j] = load_b_frag(&Bs[buf][(wn * 32 + j * 16) * 32], 32);
#pragma unroll
    for (int i = 0; i < 4; ++i)
#pragma unroll
      for (int j = 0; j < 2; ++j)
        acc[i][j] = wmma_bf16(a[i], b[j], acc[i][j]);
    if (issuer) __builtin_amdgcn_s_wait_tensorcnt(0);
    __syncthreads();
  }

  const int lane  = threadIdx.x & 31;
  const int colo  = lane & 15;
  const int rbase = (lane >> 4) << 3;
#pragma unroll
  for (int i = 0; i < 4; ++i) {
#pragma unroll
    for (int j = 0; j < 2; ++j) {
      const int n = n0 + j * 16 + colo;
      const float bias = bias0[n] + (twoBias ? bias1[n] : 0.0f);
#pragma unroll
      for (int v = 0; v < 8; ++v) {
        const int row = m0 + i * 16 + rbase + v;
        const float val = acc[i][j][v] + bias;
        if (ntStore)
          __builtin_nontemporal_store(val, &C[(size_t)row * N + n]);
        else
          C[(size_t)row * N + n] = val;
      }
    }
  }
}

// ---------------------------------------------------------------------------
// Persistent per-layer recurrence: all 256 timesteps in ONE kernel.
// 16 WGs x 8 waves = 128 waves; wave owns one 16x16 h-tile and computes its
// FOUR gate tiles per step (no intra-step cross-wave dependency). Steps are
// separated by a generation-counting agent-scope grid barrier; h ping-pongs
// across the barrier, c and accumulators are wave-exclusive. Wh (8 MB bf16)
// streams from the 192 MB L2 every step.
// ---------------------------------------------------------------------------
__global__ __launch_bounds__(256) void lstm_layer_kernel(
    const float* __restrict__ gx,      // [T*B, 4H]
    const bf16* __restrict__ Wh,       // [4H, H] (DropConnect-masked)
    bf16* __restrict__ hping,          // [2][B, H]
    float* __restrict__ c_st,          // [B, H]
    bf16* __restrict__ act_out,        // [T, B, H]
    const float* __restrict__ mask,    // [B, H]
    unsigned* __restrict__ barrier_cnt,
    float* __restrict__ hT, float* __restrict__ cT) {
  const int wid = blockIdx.x * (blockDim.x >> 5) + (threadIdx.x >> 5);  // 0..127
  const int m0 = (wid & 1) << 4;   // batch rows
  const int c0 = (wid >> 1) << 4;  // h cols
  const int lane  = threadIdx.x & 31;
  const int col   = c0 + (lane & 15);
  const int rbase = (lane >> 4) << 3;

  for (int t = 0; t < TSTEPS; ++t) {
    const bf16* h_in = hping + (size_t)(t & 1) * BATCH * HDIM;
    bf16*       h_out = hping + (size_t)((t + 1) & 1) * BATCH * HDIM;
    const float* gx_t = gx + (size_t)t * BATCH * G4H;

    v8f acc[4] = {};
    for (int k = 0; k < HDIM; k += 32) {
      const v16bf a = load_a_frag(h_in + (size_t)m0 * HDIM + k, HDIM);
#pragma unroll
      for (int g = 0; g < 4; ++g) {
        __builtin_prefetch(Wh + (size_t)(c0 + g * HDIM) * HDIM + k + 64, 0, 1);
        const v16bf b = load_b_frag(Wh + (size_t)(c0 + g * HDIM) * HDIM + k, HDIM);
        acc[g] = wmma_bf16(a, b, acc[g]);
      }
    }

#pragma unroll
    for (int v = 0; v < 8; ++v) {
      const int row = m0 + rbase + v;
      const float* gr = gx_t + (size_t)row * G4H;
      const float gi = acc[0][v] + gr[0 * HDIM + col];
      const float gf = acc[1][v] + gr[1 * HDIM + col];
      const float gg = acc[2][v] + gr[2 * HDIM + col];
      const float go = acc[3][v] + gr[3 * HDIM + col];
      const float cp = c_st[(size_t)row * HDIM + col];
      const float cn = sigmoidf_(gf) * cp + sigmoidf_(gi) * tanhf(gg);
      const float hn = sigmoidf_(go) * tanhf(cn);
      c_st[(size_t)row * HDIM + col] = cn;
      h_out[(size_t)row * HDIM + col] = (bf16)hn;
      act_out[((size_t)t * BATCH + row) * HDIM + col] =
          (bf16)(hn * mask[(size_t)row * HDIM + col]);
      if (t == TSTEPS - 1) {
        hT[(size_t)row * HDIM + col] = hn;
        cT[(size_t)row * HDIM + col] = cn;
      }
    }

    // Grid-wide step barrier (skip after the final step). Release-add makes
    // this block's h writes visible; acquire-spin pulls the other blocks'.
    if (t != TSTEPS - 1) {
      __threadfence();
      __syncthreads();
      if (threadIdx.x == 0) {
        const unsigned target = (unsigned)(t + 1) * STEP_BLOCKS;
        __hip_atomic_fetch_add(barrier_cnt, 1u, __ATOMIC_RELEASE,
                               __HIP_MEMORY_SCOPE_AGENT);
        while (__hip_atomic_load(barrier_cnt, __ATOMIC_ACQUIRE,
                                 __HIP_MEMORY_SCOPE_AGENT) < target) {
          __builtin_amdgcn_s_sleep(2);
        }
      }
      __syncthreads();
    }
  }
}

// ---------------------------------------------------------------------------
// Host driver: all launches on `stream`; workspace carved from d_ws.
// ---------------------------------------------------------------------------
extern "C" void kernel_launch(void* const* d_in, const int* in_sizes, int n_in,
                              void* d_out, int out_size, void* d_ws, size_t ws_size,
                              hipStream_t stream) {
  const int*   x      = (const int*)  d_in[0];
  const float* emb    = (const float*)d_in[1];
  const float* W_ih   = (const float*)d_in[2];
  const float* W_hh   = (const float*)d_in[3];
  const float* b_ih   = (const float*)d_in[4];
  const float* b_hh   = (const float*)d_in[5];
  const float* maskhh = (const float*)d_in[6];
  const float* h0     = (const float*)d_in[7];
  const float* c0     = (const float*)d_in[8];
  const float* m_emb  = (const float*)d_in[9];
  const float* m_mid  = (const float*)d_in[10];
  const float* m_out  = (const float*)d_in[11];
  const float* W_out  = (const float*)d_in[12];
  const float* b_out  = (const float*)d_in[13];

  char* w = (char*)d_ws;
  auto carve = [&](size_t bytes) -> char* {
    char* p = w;
    w += (bytes + 255) & ~(size_t)255;
    return p;
  };
  bf16*     actA  = (bf16*)    carve((size_t)TSTEPS * BATCH * EDIM * 2);   // 16 MB
  bf16*     actB  = (bf16*)    carve((size_t)TSTEPS * BATCH * HDIM * 2);   // 16 MB
  float*    gx    = (float*)   carve((size_t)TSTEPS * BATCH * G4H * 4);    // 128 MB
  bf16*     Wi_b  = (bf16*)    carve((size_t)LAYERS * G4H * EDIM * 2);     // 24 MB
  bf16*     Wh_b  = (bf16*)    carve((size_t)LAYERS * G4H * HDIM * 2);     // 24 MB
  bf16*     Wo_b  = (bf16*)    carve((size_t)VSZ * HDIM * 2);              // 62.5 MB
  bf16*     hping = (bf16*)    carve((size_t)2 * BATCH * HDIM * 2);
  float*    cst   = (float*)   carve((size_t)BATCH * HDIM * 4);
  unsigned* bcnt  = (unsigned*)carve(256);

  float* logits = (float*)d_out;                                   // [T,B,V]
  float* hT = logits + (size_t)TSTEPS * BATCH * VSZ;               // [L,B,H]
  float* cT = hT + (size_t)LAYERS * BATCH * HDIM;                  // [L,B,H]

  const size_t nW = (size_t)LAYERS * G4H * HDIM;
  convert_kernel<<<(unsigned)(nW / 1024), 256, 0, stream>>>(W_hh, maskhh, Wh_b, 1, nW);
  convert_kernel<<<(unsigned)(nW / 1024), 256, 0, stream>>>(W_ih, W_ih, Wi_b, 0, nW);
  convert_kernel<<<(unsigned)((size_t)VSZ * HDIM / 1024), 256, 0, stream>>>(
      W_out, W_out, Wo_b, 0, (size_t)VSZ * HDIM);

  embed_kernel<<<(unsigned)((size_t)TSTEPS * BATCH * EDIM / 1024), 256, 0, stream>>>(
      x, emb, m_emb, actA);

  for (int l = 0; l < LAYERS; ++l) {
    const bf16* ain  = (l & 1) ? actB : actA;
    bf16*       aout = (l & 1) ? actA : actB;

    // gx = act @ W_ih^T + (b_ih + b_hh): M=8192, N=4096 -> 2048 WGs
    gemm_tiled_kernel<<<2048, 256, 0, stream>>>(
        ain, Wi_b + (size_t)l * G4H * EDIM, b_ih + (size_t)l * G4H,
        b_hh + (size_t)l * G4H, 1, gx, G4H, 0);

    init_state_kernel<<<BATCH * HDIM / 256, 256, 0, stream>>>(
        h0 + (size_t)l * BATCH * HDIM, c0 + (size_t)l * BATCH * HDIM, hping, cst,
        bcnt);

    const float* mask = (l < LAYERS - 1) ? (m_mid + (size_t)l * BATCH * HDIM) : m_out;

    // Entire 256-step recurrence for this layer in one persistent kernel.
    lstm_layer_kernel<<<STEP_BLOCKS, 256, 0, stream>>>(
        gx, Wh_b + (size_t)l * G4H * HDIM, hping, cst, aout, mask, bcnt,
        hT + (size_t)l * BATCH * HDIM, cT + (size_t)l * BATCH * HDIM);
  }

  // logits = act @ W_out^T + b_out: M=8192, N=32000 -> 16000 WGs
  const bf16* finalAct = ((LAYERS - 1) & 1) ? actA : actB;
  gemm_tiled_kernel<<<16000, 256, 0, stream>>>(finalAct, Wo_b, b_out, b_out, 0,
                                               logits, VSZ, 1);
}
